// ContextLayer_58737972740832
// MI455X (gfx1250) — matrix operand order repaired and checked by reading, not compiled, not run on hardware
//
#include <hip/hip_runtime.h>
#include <hip/hip_bf16.h>

#define T_TOT  32768
#define D1C    512
#define D2C    512
#define DIN    1024
#define HID    768
#define SZ     512

typedef __bf16 bf16_t;
typedef bf16_t v8bf  __attribute__((ext_vector_type(8)));    // 16 bytes
typedef bf16_t v16bf __attribute__((ext_vector_type(16)));
typedef float  v8f   __attribute__((ext_vector_type(8)));

__device__ __forceinline__ v16bf frag_cat(v8bf lo, v8bf hi) {
    return __builtin_shufflevector(lo, hi, 0, 1, 2, 3, 4, 5, 6, 7,
                                           8, 9, 10, 11, 12, 13, 14, 15);
}

// LDS row stride: 40 bf16 = 80 bytes -> every row 16B-aligned, and row*20
// dwords mod 64 banks hits 16 distinct banks for 16 consecutive rows.
#define LDA 40

// ---------------------------------------------------------------------------
// Kernel 1:  H = concat(x1,x2) @ W_linear + b      (bf16 out, fp32 accum WMMA)
// block = 256 threads (8 waves). Block tile 256(M) x 64(N), K step 32.
// wave w -> rows [w*32, w*32+32) (2 A frags), 4 N sub-tiles -> 8 WMMA / K-step.
// ---------------------------------------------------------------------------
__global__ __launch_bounds__(256) void gemm1_kernel(
    const float* __restrict__ x1, const float* __restrict__ x2,
    const float* __restrict__ Wl, const float* __restrict__ bl,
    bf16_t* __restrict__ Hout)
{
    __shared__ __align__(16) bf16_t As[256][LDA];    // 256 x 32 used
    __shared__ __align__(16) bf16_t Bs[64][LDA];     // transposed: [col][K]

    const int tid  = threadIdx.x;
    const int lane = tid & 31;
    const int wave = tid >> 5;
    const int m0 = blockIdx.x * 256;
    const int n0 = blockIdx.y * 64;

    v8f acc[2][4];
    #pragma unroll
    for (int mt = 0; mt < 2; ++mt)
        #pragma unroll
        for (int nt = 0; nt < 4; ++nt)
            #pragma unroll
            for (int q = 0; q < 8; ++q) acc[mt][nt][q] = 0.0f;

    for (int k0 = 0; k0 < DIN; k0 += 32) {
        // ---- stage A: one 32-wide row slice per thread (K tile never
        // straddles the x1/x2 boundary since 512 % 32 == 0)
        {
            const float* src = (k0 < D1C)
                ? (x1 + (size_t)(m0 + tid) * D1C + k0)
                : (x2 + (size_t)(m0 + tid) * D2C + (k0 - D1C));
            #pragma unroll
            for (int e = 0; e < 32; e += 8) {
                v8bf v;
                #pragma unroll
                for (int q = 0; q < 8; ++q) v[q] = (bf16_t)src[e + q];
                *(v8bf*)&As[tid][e] = v;
            }
        }
        // ---- stage B transposed: thread reads 8 consecutive cols of one K row
        {
            const int kk  = (tid * 8) >> 6;
            const int nn0 = (tid * 8) & 63;
            const float* bsrc = Wl + (size_t)(k0 + kk) * HID + (n0 + nn0);
            #pragma unroll
            for (int q = 0; q < 8; ++q) Bs[nn0 + q][kk] = (bf16_t)bsrc[q];
        }
        __syncthreads();

        // ---- A fragments (ISA 7.12.2): row = base + lane%16,
        // lanes 0-15 K {0..7,16..23}, lanes 16-31 K {8..15,24..31}
        const int kb = (lane >> 4) * 8;
        v16bf a[2];
        #pragma unroll
        for (int mt = 0; mt < 2; ++mt) {
            const int ar = wave * 32 + mt * 16 + (lane & 15);
            v8bf lo = *(const v8bf*)&As[ar][kb];
            v8bf hi = *(const v8bf*)&As[ar][kb + 16];
            a[mt] = frag_cat(lo, hi);
        }
        #pragma unroll
        for (int nt = 0; nt < 4; ++nt) {
            // B fragment: col = lane%16; lanes 0-15 K {0..7,16..23}, 16-31 K {8..15,24..31}
            const int bc  = nt * 16 + (lane & 15);
            const int kbB = (lane >> 4) * 8;
            v8bf lo = *(const v8bf*)&Bs[bc][kbB];
            v8bf hi = *(const v8bf*)&Bs[bc][kbB + 16];
            v16bf b = frag_cat(lo, hi);
            #pragma unroll
            for (int mt = 0; mt < 2; ++mt)
                acc[mt][nt] = __builtin_amdgcn_wmma_f32_16x16x32_bf16(
                    false, a[mt], false, b, (short)0, acc[mt][nt], false, false);
        }
        __syncthreads();
    }

    // epilogue: +bias, store bf16. C layout: VGPR v -> M = v + 8*(lane>=16)
    #pragma unroll
    for (int mt = 0; mt < 2; ++mt)
        #pragma unroll
        for (int nt = 0; nt < 4; ++nt)
            #pragma unroll
            for (int v = 0; v < 8; ++v) {
                int m = m0 + wave * 32 + mt * 16 + v + ((lane >> 4) * 8);
                int n = n0 + nt * 16 + (lane & 15);
                Hout[(size_t)m * HID + n] = (bf16_t)(acc[mt][nt][v] + bl[n]);
            }
}

// ---------------------------------------------------------------------------
// Kernel 2:  Z = sigmoid(H @ W_embed)   (bf16 in, bf16 out)
// ---------------------------------------------------------------------------
__global__ __launch_bounds__(256) void gemm2_kernel(
    const bf16_t* __restrict__ H, const float* __restrict__ We,
    bf16_t* __restrict__ Zout)
{
    __shared__ __align__(16) bf16_t As[256][LDA];
    __shared__ __align__(16) bf16_t Bs[64][LDA];

    const int tid  = threadIdx.x;
    const int lane = tid & 31;
    const int wave = tid >> 5;
    const int m0 = blockIdx.x * 256;
    const int n0 = blockIdx.y * 64;

    v8f acc[2][4];
    #pragma unroll
    for (int mt = 0; mt < 2; ++mt)
        #pragma unroll
        for (int nt = 0; nt < 4; ++nt)
            #pragma unroll
            for (int q = 0; q < 8; ++q) acc[mt][nt][q] = 0.0f;

    for (int k0 = 0; k0 < HID; k0 += 32) {
        // ---- stage A: straight bf16 copy, 4 x 16B per thread
        {
            const v8bf* srcv = (const v8bf*)(H + (size_t)(m0 + tid) * HID + k0);
            #pragma unroll
            for (int e = 0; e < 4; ++e) *(v8bf*)&As[tid][e * 8] = srcv[e];
        }
        // ---- stage B transposed
        {
            const int kk  = (tid * 8) >> 6;
            const int nn0 = (tid * 8) & 63;
            const float* bsrc = We + (size_t)(k0 + kk) * SZ + (n0 + nn0);
            #pragma unroll
            for (int q = 0; q < 8; ++q) Bs[nn0 + q][kk] = (bf16_t)bsrc[q];
        }
        __syncthreads();

        const int kb = (lane >> 4) * 8;
        v16bf a[2];
        #pragma unroll
        for (int mt = 0; mt < 2; ++mt) {
            const int ar = wave * 32 + mt * 16 + (lane & 15);
            v8bf lo = *(const v8bf*)&As[ar][kb];
            v8bf hi = *(const v8bf*)&As[ar][kb + 16];
            a[mt] = frag_cat(lo, hi);
        }
        #pragma unroll
        for (int nt = 0; nt < 4; ++nt) {
            const int bc  = nt * 16 + (lane & 15);
            const int kbB = (lane >> 4) * 8;
            v8bf lo = *(const v8bf*)&Bs[bc][kbB];
            v8bf hi = *(const v8bf*)&Bs[bc][kbB + 16];
            v16bf b = frag_cat(lo, hi);
            #pragma unroll
            for (int mt = 0; mt < 2; ++mt)
                acc[mt][nt] = __builtin_amdgcn_wmma_f32_16x16x32_bf16(
                    false, a[mt], false, b, (short)0, acc[mt][nt], false, false);
        }
        __syncthreads();
    }

    #pragma unroll
    for (int mt = 0; mt < 2; ++mt)
        #pragma unroll
        for (int nt = 0; nt < 4; ++nt)
            #pragma unroll
            for (int v = 0; v < 8; ++v) {
                int m = m0 + wave * 32 + mt * 16 + v + ((lane >> 4) * 8);
                int n = n0 + nt * 16 + (lane & 15);
                float z = 1.0f / (1.0f + __expf(-acc[mt][nt][v]));
                Zout[(size_t)m * SZ + n] = (bf16_t)z;
            }
}

// ---------------------------------------------------------------------------
// Kernel 3:  r[t] = tanh(Z[t,:] @ W_rate)   — one wave32 per row
// ---------------------------------------------------------------------------
__global__ __launch_bounds__(256) void rate_kernel(
    const bf16_t* __restrict__ Z, const float* __restrict__ Wr,
    float* __restrict__ r)
{
    const int lane = threadIdx.x & 31;
    const int wid  = threadIdx.x >> 5;
    const int t    = blockIdx.x * 8 + wid;
    const v8bf* zr = (const v8bf*)(Z + (size_t)t * SZ + lane * 16);
    v8bf z0 = zr[0], z1 = zr[1];
    float acc = 0.0f;
    #pragma unroll
    for (int e = 0; e < 8; ++e) {
        acc += (float)z0[e] * Wr[lane * 16 + e];
        acc += (float)z1[e] * Wr[lane * 16 + 8 + e];
    }
    #pragma unroll
    for (int off = 16; off > 0; off >>= 1) acc += __shfl_xor(acc, off, 32);
    if (lane == 0) r[t] = tanhf(acc);
}

// ---------------------------------------------------------------------------
// Kernel 4: sequential recurrence. Single persistent workgroup (1024 thr = 32
// waves). c[512] in LDS; W_forget (1MB fp32) streams from 192MB L2 each step
// as float2 column-pair loads. 1024 threads = 256 col-pairs x 4 K-quarters
// (serial chain of 128 FMAs). Softmax via wave32 shuffles + one LDS stage.
// ---------------------------------------------------------------------------
__global__ __launch_bounds__(1024) void recurrence_kernel(
    const float* __restrict__ Wf, const bf16_t* __restrict__ Zbf,
    const float* __restrict__ r, const float* __restrict__ c0,
    float* __restrict__ out, int T)
{
    __shared__ float  c[SZ];
    __shared__ float2 partial2[1024];
    __shared__ float  logits[SZ];
    __shared__ float  wred[32];
    __shared__ float  m_s, s_s;

    const int tid  = threadIdx.x;
    const int p    = tid & 255;     // column pair index (j = 2p, 2p+1)
    const int q    = tid >> 8;      // K quarter (128 each)
    const int lane = tid & 31;
    const int wid  = tid >> 5;

    if (tid < SZ) c[tid] = c0[tid];
    __syncthreads();

    for (int t = 0; t < T; ++t) {
        // prefetch next z row (gfx1250 global_prefetch_b8)
        if (t + 1 < T && tid < SZ)
            __builtin_prefetch(&Zbf[(size_t)(t + 1) * SZ + tid], 0, 0);

        // ---- partial matvec: float2 loads over one K quarter
        float ax = 0.0f, ay = 0.0f;
        const int kbase = q * 128;
        const float* Wp = Wf + (size_t)kbase * SZ + p * 2;
        #pragma unroll 8
        for (int k = 0; k < 128; ++k) {
            float ck = c[kbase + k];
            float2 w = *(const float2*)(Wp + (size_t)k * SZ);
            ax = fmaf(ck, w.x, ax);
            ay = fmaf(ck, w.y, ay);
        }
        partial2[tid] = make_float2(ax, ay);
        __syncthreads();

        // ---- combine quarters + max reduce
        float m_local = -1e30f;
        float e = 0.0f;
        if (tid < SZ) {
            const int pp = tid >> 1;
            float2 s0 = partial2[pp];
            float2 s1 = partial2[256 + pp];
            float2 s2 = partial2[512 + pp];
            float2 s3 = partial2[768 + pp];
            float l = (tid & 1) ? (s0.y + s1.y + s2.y + s3.y)
                                : (s0.x + s1.x + s2.x + s3.x);
            logits[tid] = l;
            m_local = l;
        }
        #pragma unroll
        for (int off = 16; off > 0; off >>= 1)
            m_local = fmaxf(m_local, __shfl_xor(m_local, off, 32));
        if (lane == 0) wred[wid] = m_local;
        __syncthreads();
        if (wid == 0) {
            float v = wred[lane];
            #pragma unroll
            for (int off = 16; off > 0; off >>= 1)
                v = fmaxf(v, __shfl_xor(v, off, 32));
            if (lane == 0) m_s = v;
        }
        __syncthreads();

        // ---- exp + sum reduce
        if (tid < SZ) e = __expf(logits[tid] - m_s);
        float s_local = e;
        #pragma unroll
        for (int off = 16; off > 0; off >>= 1)
            s_local += __shfl_xor(s_local, off, 32);
        if (lane == 0) wred[wid] = s_local;
        __syncthreads();
        if (wid == 0) {
            float v = wred[lane];
            #pragma unroll
            for (int off = 16; off > 0; off >>= 1) v += __shfl_xor(v, off, 32);
            if (lane == 0) s_s = v;
        }
        __syncthreads();

        // ---- state update
        if (tid < SZ) {
            float zt = (float)Zbf[(size_t)t * SZ + tid];
            float rt = r[t];
            float cn = (e / s_s) * c[tid] + rt * zt;
            c[tid] = cn;
            out[(size_t)t * SZ + tid] = cn;
        }
        __syncthreads();
    }
}

// ---------------------------------------------------------------------------
extern "C" void kernel_launch(void* const* d_in, const int* in_sizes, int n_in,
                              void* d_out, int out_size, void* d_ws, size_t ws_size,
                              hipStream_t stream)
{
    const float* x1 = (const float*)d_in[0];
    const float* x2 = (const float*)d_in[1];
    const float* c0 = (const float*)d_in[2];
    const float* Wl = (const float*)d_in[3];
    const float* bl = (const float*)d_in[4];
    const float* We = (const float*)d_in[5];
    const float* Wr = (const float*)d_in[6];
    const float* Wf = (const float*)d_in[7];
    float* out = (float*)d_out;

    const int T = T_TOT;
    char* ws = (char*)d_ws;
    bf16_t* Hbf = (bf16_t*)ws;                                      // T*HID bf16 (48MB)
    bf16_t* Zbf = (bf16_t*)(ws + (size_t)T * HID * 2);              // T*SZ  bf16 (32MB)
    float*  rr  = (float*)(ws + (size_t)T * HID * 2 + (size_t)T * SZ * 2); // T fp32

    gemm1_kernel<<<dim3(T / 256, HID / 64), dim3(256), 0, stream>>>(x1, x2, Wl, bl, Hbf);
    gemm2_kernel<<<dim3(T / 256, SZ / 64),  dim3(256), 0, stream>>>(Hbf, We, Zbf);
    rate_kernel <<<dim3(T / 8),             dim3(256), 0, stream>>>(Zbf, Wr, rr);
    recurrence_kernel<<<1, 1024, 0, stream>>>(Wf, Zbf, rr, c0, out, T);
}
